// MPiFormerPointNet_65111704207595
// MI455X (gfx1250) — matrix operand order repaired and checked by reading, not compiled, not run on hardware
//
#include <hip/hip_runtime.h>

typedef __attribute__((ext_vector_type(16))) _Float16 v16h;
typedef __attribute__((ext_vector_type(8)))  _Float16 v8h;
typedef __attribute__((ext_vector_type(8)))  float    v8f;

#define B_   8
#define N_   4096
#define FIN  32
#define NRP_ 1024
#define KNB  64

// ---------- order-preserving float<->uint encoding for atomic max ----------
__device__ __forceinline__ unsigned encf(float f) {
    unsigned u = __float_as_uint(f);
    return (u & 0x80000000u) ? ~u : (u | 0x80000000u);
}
__device__ __forceinline__ float decf(unsigned k) {
    return __uint_as_float((k & 0x80000000u) ? (k & 0x7fffffffu) : ~k);
}

// ---------- embedding: x0[b,n,:] = emb_table[feat[b,n]] (+ point_id_emb for n<1024) ----------
__global__ void embed_kernel(const int* __restrict__ feat, const float* __restrict__ emb,
                             const float* __restrict__ pid, float* __restrict__ x0, int total) {
    int t = blockIdx.x * blockDim.x + threadIdx.x;
    if (t >= total) return;
    int c = t & (FIN - 1);
    int n = (t / FIN) % N_;
    int b = t / (FIN * N_);
    int e = feat[b * N_ + n];
    float v = emb[e * FIN + c];
    if (n < NRP_) v += pid[n * FIN + c];
    x0[t] = v;
}

// ---------- farthest point sampling (one block per batch, matches jnp.argmax tie rule) ----------
__global__ void __launch_bounds__(256) fps_kernel(const float* __restrict__ pos, int Nin, int ms,
                                                  int* __restrict__ idx, float* __restrict__ pos_s) {
    __shared__ float dist[4096];
    __shared__ float bv[256];
    __shared__ int   bi[256];
    __shared__ float curP[3];
    const int b = blockIdx.x, tid = threadIdx.x;
    const float* base = pos + (size_t)b * Nin * 3;
    const float p0x = base[0], p0y = base[1], p0z = base[2];
    for (int n = tid; n < Nin; n += 256) {
        float dx = base[n*3+0] - p0x, dy = base[n*3+1] - p0y, dz = base[n*3+2] - p0z;
        dist[n] = dx*dx + dy*dy + dz*dz;
    }
    if (tid == 0) {
        idx[(size_t)b * ms] = 0;
        pos_s[(size_t)b * ms * 3 + 0] = p0x;
        pos_s[(size_t)b * ms * 3 + 1] = p0y;
        pos_s[(size_t)b * ms * 3 + 2] = p0z;
    }
    __syncthreads();
    for (int it = 1; it < ms; ++it) {
        float val = -1.0f; int best = 0x7fffffff;
        for (int n = tid; n < Nin; n += 256) {
            float v = dist[n];
            if (v > val) { val = v; best = n; }
        }
        bv[tid] = val; bi[tid] = best;
        __syncthreads();
        for (int s = 128; s > 0; s >>= 1) {
            if (tid < s) {
                if (bv[tid+s] > bv[tid] || (bv[tid+s] == bv[tid] && bi[tid+s] < bi[tid])) {
                    bv[tid] = bv[tid+s]; bi[tid] = bi[tid+s];
                }
            }
            __syncthreads();
        }
        if (tid == 0) {
            int nx = bi[0];
            idx[(size_t)b * ms + it] = nx;
            curP[0] = base[nx*3+0]; curP[1] = base[nx*3+1]; curP[2] = base[nx*3+2];
            pos_s[((size_t)b * ms + it) * 3 + 0] = curP[0];
            pos_s[((size_t)b * ms + it) * 3 + 1] = curP[1];
            pos_s[((size_t)b * ms + it) * 3 + 2] = curP[2];
        }
        __syncthreads();
        const float cx = curP[0], cy = curP[1], cz = curP[2];
        for (int n = tid; n < Nin; n += 256) {
            float dx = base[n*3+0] - cx, dy = base[n*3+1] - cy, dz = base[n*3+2] - cz;
            dist[n] = fminf(dist[n], dx*dx + dy*dy + dz*dz);
        }
        __syncthreads();
    }
}

// ---------- radius-limited K nearest neighbors (one block per sampled point) ----------
// ms is a power of two; msShift = log2(ms).
__global__ void __launch_bounds__(256) radius_kernel(const float* __restrict__ pos,
                                                     const float* __restrict__ pos_s,
                                                     int Nin, int ms, int msShift, float r2,
                                                     int* __restrict__ nbr) {
    __shared__ float d2s[4096];
    __shared__ float bv[256];
    __shared__ int   bi[256];
    const int blk = blockIdx.x, tid = threadIdx.x;
    const int i = blk & (ms - 1), b = blk >> msShift;
    const float* base = pos + (size_t)b * Nin * 3;
    const float sx = pos_s[((size_t)b*ms + i)*3 + 0];
    const float sy = pos_s[((size_t)b*ms + i)*3 + 1];
    const float sz = pos_s[((size_t)b*ms + i)*3 + 2];
    for (int n = tid; n < Nin; n += 256) {
        float dx = base[n*3+0] - sx, dy = base[n*3+1] - sy, dz = base[n*3+2] - sz;
        d2s[n] = dx*dx + dy*dy + dz*dz;
    }
    __syncthreads();
    int* out = nbr + ((size_t)b*ms + i) * KNB;
    for (int k = 0; k < KNB; ++k) {
        float val = 3.0e38f; int best = 0x7fffffff;
        for (int n = tid; n < Nin; n += 256) {
            float v = d2s[n];
            if (v < val) { val = v; best = n; }
        }
        bv[tid] = val; bi[tid] = best;
        __syncthreads();
        for (int s = 128; s > 0; s >>= 1) {
            if (tid < s) {
                if (bv[tid+s] < bv[tid] || (bv[tid+s] == bv[tid] && bi[tid+s] < bi[tid])) {
                    bv[tid] = bv[tid+s]; bi[tid] = bi[tid+s];
                }
            }
            __syncthreads();
        }
        if (tid == 0) {
            if (bv[0] <= r2) { out[k] = bi[0]; d2s[bi[0]] = 3.0e38f; }
            else             { out[k] = -1; }
        }
        __syncthreads();
    }
}

__global__ void fill_kernel(unsigned* __restrict__ p, unsigned v, int n) {
    int t = blockIdx.x * blockDim.x + threadIdx.x;
    if (t < n) p[t] = v;
}

__global__ void decode_kernel(const unsigned* __restrict__ agg, float* __restrict__ out, int n) {
    int t = blockIdx.x * blockDim.x + threadIdx.x;
    if (t < n) out[t] = decf(agg[t]);
}

__global__ void final_kernel(const unsigned* __restrict__ agg, const float* __restrict__ ps,
                             float* __restrict__ out, int nx, int np) {
    int t = blockIdx.x * blockDim.x + threadIdx.x;
    if (t < nx)            out[t] = decf(agg[t]);
    else if (t < nx + np)  out[t] = ps[t - nx];
}

// ---------- one MLP layer via WMMA: out(16 x Nout) = act(in(16 x KinPad) @ W + b) ----------
// in/out are f16 in LDS, W/bias f32 in global. Nout is a multiple of 64 so the N-chunk is
// always 64 or 128 (power of two -> mask/shift indexing, no integer division).
// If aggBase != nullptr this is the final layer: encoded results are atomicMax'ed into agg.
__device__ __forceinline__ void run_layer(const float* __restrict__ W, const float* __restrict__ bias,
                                          int KinLogical, int KinPad, int Nout,
                                          const _Float16* in, _Float16* out, _Float16* wt,
                                          bool relu, unsigned* aggBase,
                                          const int* nbrIdx, int tid) {
    const int lane  = tid & 31;
    const int wave  = tid >> 5;
    const int nlane = lane & 15;
    const int hi    = (lane >= 16) ? 1 : 0;
    for (int nc = 0; nc < Nout; nc += 128) {
        const int chunk  = (Nout - nc < 128) ? (Nout - nc) : 128;   // 64 or 128
        const int cshift = (chunk == 128) ? 7 : 6;
        const int cmask  = chunk - 1;
        const int ntiles = chunk >> 4;
        v8f acc = {};
        for (int k0 = 0; k0 < KinPad; k0 += 32) {
            __syncthreads();
            // stage transposed weight tile wt[n_local][k_local], f32 -> f16
            for (int q = tid; q < (chunk << 5); q += 256) {
                int nl = q & cmask;
                int kk = q >> cshift;
                int k = k0 + kk;
                float w = (k < KinLogical) ? W[(size_t)k * Nout + nc + nl] : 0.0f;
                wt[(nl << 5) + kk] = (_Float16)w;
            }
            // prefetch next k-step's weight rows into cache (global_prefetch_b8)
            {
                int kp = k0 + 32 + (tid & 31);
                if (kp < KinLogical)
                    __builtin_prefetch(&W[(size_t)kp * Nout + nc + ((tid >> 5) << 4)], 0, 3);
            }
            __syncthreads();
            if (wave < ntiles) {
                // A fragment: M = lane%16; halves {0..7}=K[kb1..kb1+7], {8..15}=K[kb2..kb2+7]
                const _Float16* arow = in + nlane * KinPad + k0;
                v8h alo = *(const v8h*)(arow + (hi ? 8 : 0));
                v8h ahi = *(const v8h*)(arow + (hi ? 24 : 16));
                v16h a = __builtin_shufflevector(alo, ahi, 0,1,2,3,4,5,6,7,8,9,10,11,12,13,14,15);
                // B fragment: N = tile*16 + lane%16; halves {0..7}=K[kb1..kb1+7], {8..15}=K[kb2..kb2+7]
                const _Float16* brow = wt + ((wave * 16 + nlane) << 5) + (hi ? 8 : 0);
                v8h blo = *(const v8h*)(brow);
                v8h bhi = *(const v8h*)(brow + 16);
                v16h bf = __builtin_shufflevector(blo, bhi, 0,1,2,3,4,5,6,7,8,9,10,11,12,13,14,15);
                acc = __builtin_amdgcn_wmma_f32_16x16x32_f16(false, a, false, bf,
                                                             (short)0, acc, false, false);
            }
        }
        __syncthreads();
        if (wave < ntiles) {
            const int n = nc + wave * 16 + nlane;
            const float bvv = bias[n];
#pragma unroll
            for (int v = 0; v < 8; ++v) {
                float r = acc[v] + bvv;
                if (relu) r = fmaxf(r, 0.0f);
                const int mrow = v + (hi ? 8 : 0);
                if (aggBase) {
                    if (nbrIdx[mrow] >= 0) atomicMax(&aggBase[n], encf(r));
                } else {
                    out[mrow * Nout + n] = (_Float16)r;
                }
            }
        }
    }
}

// ---------- fused SA-stage conv: gather -> 3-layer MLP (WMMA) -> masked max (atomic) ----------
// grid = B * ms * (KNB/16); each block handles 16 neighbor rows of one sampled point.
// ms is a power of two; msShift = log2(ms).
__global__ void __launch_bounds__(256) conv_stage_kernel(
    const float* __restrict__ x, int F, const float* __restrict__ pos,
    const float* __restrict__ pos_s, const int* __restrict__ nbr,
    int Nin, int ms, int msShift,
    const float* __restrict__ W0, const float* __restrict__ B0,
    const float* __restrict__ W1, const float* __restrict__ B1,
    const float* __restrict__ W2, const float* __restrict__ B2,
    int Cin, int Kpad0, int C1, int C2, int C3,
    unsigned* __restrict__ agg) {
    __shared__ __align__(16) _Float16 bufA[16 * 512];
    __shared__ __align__(16) _Float16 bufB[16 * 512];
    __shared__ __align__(16) _Float16 wt[128 * 32];
    __shared__ int   nbrIdx[16];
    __shared__ float psLoc[3];
    const int tid  = threadIdx.x;
    const int blk  = blockIdx.x;
    const int tile = blk & 3;
    const int rest = blk >> 2;
    const int i = rest & (ms - 1);
    const int b = rest >> msShift;

    const int* nrow = nbr + ((size_t)b * ms + i) * KNB + tile * 16;
    if (tid < 16) nbrIdx[tid] = nrow[tid];
    if (tid < 3)  psLoc[tid] = pos_s[((size_t)b * ms + i) * 3 + tid];
    __syncthreads();

    // gather: row = [x_j (F) | p_j - pos_s (3) | zero pad]; outer row loop avoids division
    for (int mr = 0; mr < 16; ++mr) {
        const int j = nbrIdx[mr];
        const float* xrow = x + ((size_t)b * Nin + (j < 0 ? 0 : j)) * F;
        const float* prow = pos + ((size_t)b * Nin + (j < 0 ? 0 : j)) * 3;
        for (int c = tid; c < Kpad0; c += 256) {
            float v = 0.0f;
            if (j >= 0) {
                if (c < F)          v = xrow[c];
                else if (c < F + 3) v = prow[c - F] - psLoc[c - F];
            }
            bufA[mr * Kpad0 + c] = (_Float16)v;
        }
    }
    __syncthreads();

    run_layer(W0, B0, Cin, Kpad0, C1, bufA, bufB, wt, true,  nullptr, nbrIdx, tid);
    __syncthreads();
    run_layer(W1, B1, C1,  C1,    C2, bufB, bufA, wt, true,  nullptr, nbrIdx, tid);
    __syncthreads();
    run_layer(W2, B2, C2,  C2,    C3, bufA, bufB, wt, false,
              agg + ((size_t)b * ms + i) * C3, nbrIdx, tid);
}

extern "C" void kernel_launch(void* const* d_in, const int* in_sizes, int n_in,
                              void* d_out, int out_size, void* d_ws, size_t ws_size,
                              hipStream_t stream) {
    (void)in_sizes; (void)n_in; (void)out_size; (void)ws_size;
    const int*   feat = (const int*)d_in[0];
    const float* pos0 = (const float*)d_in[1];
    const float* emb  = (const float*)d_in[2];
    const float* pid  = (const float*)d_in[3];
    const float* s1w0 = (const float*)d_in[4];  const float* s1b0 = (const float*)d_in[5];
    const float* s1w1 = (const float*)d_in[6];  const float* s1b1 = (const float*)d_in[7];
    const float* s1w2 = (const float*)d_in[8];  const float* s1b2 = (const float*)d_in[9];
    const float* s2w0 = (const float*)d_in[10]; const float* s2b0 = (const float*)d_in[11];
    const float* s2w1 = (const float*)d_in[12]; const float* s2b1 = (const float*)d_in[13];
    const float* s2w2 = (const float*)d_in[14]; const float* s2b2 = (const float*)d_in[15];
    const float* s3w0 = (const float*)d_in[16]; const float* s3b0 = (const float*)d_in[17];
    const float* s3w1 = (const float*)d_in[18]; const float* s3b1 = (const float*)d_in[19];
    const float* s3w2 = (const float*)d_in[20]; const float* s3b2 = (const float*)d_in[21];

    char* ws = (char*)d_ws;
    size_t off = 0;
    auto alloc = [&](size_t bytes) -> void* {
        off = (off + 255) & ~(size_t)255;
        void* p = ws + off;
        off += bytes;
        return p;
    };
    float*    x0   = (float*)   alloc((size_t)B_ * N_ * FIN * 4);
    int*      idx1 = (int*)     alloc((size_t)B_ * 1024 * 4);
    float*    ps1  = (float*)   alloc((size_t)B_ * 1024 * 3 * 4);
    int*      nbr1 = (int*)     alloc((size_t)B_ * 1024 * KNB * 4);
    unsigned* agg1 = (unsigned*)alloc((size_t)B_ * 1024 * 64 * 4);
    float*    xs1  = (float*)   alloc((size_t)B_ * 1024 * 64 * 4);
    int*      idx2 = (int*)     alloc((size_t)B_ * 256 * 4);
    float*    ps2  = (float*)   alloc((size_t)B_ * 256 * 3 * 4);
    int*      nbr2 = (int*)     alloc((size_t)B_ * 256 * KNB * 4);
    unsigned* agg2 = (unsigned*)alloc((size_t)B_ * 256 * 256 * 4);
    float*    xs2  = (float*)   alloc((size_t)B_ * 256 * 256 * 4);
    int*      idx3 = (int*)     alloc((size_t)B_ * 64 * 4);
    float*    ps3  = (float*)   alloc((size_t)B_ * 64 * 3 * 4);
    int*      nbr3 = (int*)     alloc((size_t)B_ * 64 * KNB * 4);
    unsigned* agg3 = (unsigned*)alloc((size_t)B_ * 64 * 512 * 4);

    // embedding
    {
        int total = B_ * N_ * FIN;
        embed_kernel<<<(total + 255) / 256, 256, 0, stream>>>(feat, emb, pid, x0, total);
    }
    // ---- stage 1: m=1024 (shift 10), r=0.05 ----
    fps_kernel<<<B_, 256, 0, stream>>>(pos0, N_, 1024, idx1, ps1);
    radius_kernel<<<B_ * 1024, 256, 0, stream>>>(pos0, ps1, N_, 1024, 10, 0.05f * 0.05f, nbr1);
    {
        int n = B_ * 1024 * 64;
        fill_kernel<<<(n + 255) / 256, 256, 0, stream>>>(agg1, 0u, n);
    }
    conv_stage_kernel<<<B_ * 1024 * 4, 256, 0, stream>>>(
        x0, FIN, pos0, ps1, nbr1, N_, 1024, 10,
        s1w0, s1b0, s1w1, s1b1, s1w2, s1b2,
        35, 64, 64, 64, 64, agg1);
    {
        int n = B_ * 1024 * 64;
        decode_kernel<<<(n + 255) / 256, 256, 0, stream>>>(agg1, xs1, n);
    }
    // ---- stage 2: m=256 (shift 8), r=0.3 ----
    fps_kernel<<<B_, 256, 0, stream>>>(ps1, 1024, 256, idx2, ps2);
    radius_kernel<<<B_ * 256, 256, 0, stream>>>(ps1, ps2, 1024, 256, 8, 0.3f * 0.3f, nbr2);
    {
        int n = B_ * 256 * 256;
        fill_kernel<<<(n + 255) / 256, 256, 0, stream>>>(agg2, 0u, n);
    }
    conv_stage_kernel<<<B_ * 256 * 4, 256, 0, stream>>>(
        xs1, 64, ps1, ps2, nbr2, 1024, 256, 8,
        s2w0, s2b0, s2w1, s2b1, s2w2, s2b2,
        67, 96, 128, 128, 256, agg2);
    {
        int n = B_ * 256 * 256;
        decode_kernel<<<(n + 255) / 256, 256, 0, stream>>>(agg2, xs2, n);
    }
    // ---- stage 3: m=64 (shift 6), r=0.5 ----
    fps_kernel<<<B_, 256, 0, stream>>>(ps2, 256, 64, idx3, ps3);
    radius_kernel<<<B_ * 64, 256, 0, stream>>>(ps2, ps3, 256, 64, 6, 0.5f * 0.5f, nbr3);
    {
        int n = B_ * 64 * 512;
        fill_kernel<<<(n + 255) / 256, 256, 0, stream>>>(agg3, 0u, n);
    }
    conv_stage_kernel<<<B_ * 64 * 4, 256, 0, stream>>>(
        xs2, 256, ps2, ps3, nbr3, 256, 64, 6,
        s3w0, s3b0, s3w1, s3b1, s3w2, s3b2,
        259, 288, 256, 512, 512, agg3);
    // ---- output: x (B,64,512) then pos (B,64,3) ----
    {
        int nx = B_ * 64 * 512;
        int np = B_ * 64 * 3;
        final_kernel<<<(nx + np + 255) / 256, 256, 0, stream>>>(agg3, ps3, (float*)d_out, nx, np);
    }
}
